// Spatial_Attention_38465727103730
// MI455X (gfx1250) — matrix operand order repaired and checked by reading, not compiled, not run on hardware
//
#include <hip/hip_runtime.h>
#include <hip/hip_bf16.h>
#include <math.h>

// ---------------------------------------------------------------------------
// MI455X (gfx1250) implementation of the Spatial-Attention reference.
// bf16 WMMA (v_wmma_f32_16x16x32_bf16) for all GEMMs, f32 accumulate.
// Pipeline: qkv -> E=qk^T -> (E@G + softmax, in-place, LDS-staged) ->
//           colsum(t=5) -> select y -> mixup(new_value, transposed bf16) ->
//           attn@V -> FFN1+GELU -> FFN2+residual+LayerNorm.
// LDS staging uses GLOBAL_LOAD_ASYNC_TO_LDS_B128 (ASYNCcnt-tracked DMA).
// ---------------------------------------------------------------------------

typedef __attribute__((ext_vector_type(16))) __bf16 v16bf;
typedef __attribute__((ext_vector_type(8)))  __bf16 v8bf;
typedef __attribute__((ext_vector_type(8)))  float  v8f;
typedef int v4i __attribute__((vector_size(16)));

#define BDIM   4
#define TDIM   12
#define NDIM   1024
#define DDIM   128
#define BT     48
#define CFRONT 78
#define XLAST  5
#define BSST   6

#if defined(__AMDGCN__) && __has_builtin(__builtin_amdgcn_global_load_async_to_lds_b128)
#define HAVE_ASYNC_LDS 1
#endif

// ---------------- WMMA helpers (wave32) ----------------

__device__ __forceinline__ v8f vzero8() {
  v8f z;
#pragma unroll
  for (int i = 0; i < 8; ++i) z[i] = 0.f;
  return z;
}

__device__ __forceinline__ v8f wmma_bf16(v16bf a, v16bf b, v8f c) {
  // (neg_a, A, neg_b, B, c_mod, C, reuse_a, reuse_b)
  return __builtin_amdgcn_wmma_f32_16x16x32_bf16(false, a, false, b, (short)0, c,
                                                 false, false);
}

// A-matrix fragment (16x32 bf16, MxK): lane L holds row M = L&15;
// K = (L>>4)*8 + i for i<8 and + 16 + (i-8) for i>=8.
__device__ __forceinline__ v16bf fragA_b16(const __bf16* p, int ld, int lane) {
  const int r  = lane & 15;
  const int ko = (lane >> 4) << 3;
  const __bf16* q = p + (size_t)r * ld + ko;
  union { v16bf v; v8bf h[2]; } u;
  u.h[0] = *(const v8bf*)(q);
  u.h[1] = *(const v8bf*)(q + 16);
  return u.v;
}

// Same A layout, but source is f32 (converted on the fly).
__device__ __forceinline__ v16bf fragA_f32(const float* p, int ld, int lane) {
  const int r  = lane & 15;
  const int ko = (lane >> 4) << 3;
  const float* q = p + (size_t)r * ld + ko;
  v16bf v;
#pragma unroll
  for (int i = 0; i < 8; ++i) {
    v[i]     = (__bf16)q[i];
    v[i + 8] = (__bf16)q[i + 16];
  }
  return v;
}

// B-matrix fragment (32x16 bf16, KxN): lane L holds column N = L&15;
// K = (L>>4)*8 + i for i<8 and + 16 + (i-8) for i>=8.  `p` points into TRANSPOSED storage (row = output column,
// K contiguous), K-offset already applied by caller.
__device__ __forceinline__ v16bf fragB_b16(const __bf16* p, int ld, int lane) {
  const int nco = lane & 15;
  const int kb0 = (lane >> 4) << 3;
  const __bf16* q = p + (size_t)nco * ld + kb0;
  union { v16bf v; v8bf h[2]; } u;
  u.h[0] = *(const v8bf*)(q);
  u.h[1] = *(const v8bf*)(q + 16);
  return u.v;
}

// C/D tile (16x16 f32): lane L holds column N = L&15, rows M = j + (L>>4)*8.

// ---------------- LDS staging: 16 rows x 1024 bf16 (32 KB) ----------------
// Async DMA path (no VGPR round-trip, tracked by ASYNCcnt) when available.
__device__ __forceinline__ void stage16rows(const __bf16* gsrc, __bf16* lds,
                                            int tid) {
#if defined(HAVE_ASYNC_LDS)
#pragma unroll
  for (int it = 0; it < 8; ++it) {
    const int c = it * 256 + tid;        // 2048 x 16-byte chunks
    __builtin_amdgcn_global_load_async_to_lds_b128(
        (v4i __attribute__((address_space(1)))*)(void*)(gsrc + (size_t)c * 8),
        (v4i __attribute__((address_space(3)))*)(void*)(lds + (size_t)c * 8),
        0, 0);
  }
#if __has_builtin(__builtin_amdgcn_s_wait_asynccnt)
  __builtin_amdgcn_s_wait_asynccnt(0);
#else
  asm volatile("s_wait_asynccnt 0" ::: "memory");
#endif
#else
  for (int c = tid; c < (16 * NDIM) / 8; c += 256)
    ((v8bf*)lds)[c] = ((const v8bf*)gsrc)[c];
#endif
}

// ---------------- small conversion kernels ----------------

__global__ void k_cvt_bf16(const float* __restrict__ src, __bf16* __restrict__ dst,
                           int n) {
  int g = blockIdx.x * 256 + threadIdx.x;
  if (g < n) dst[g] = (__bf16)src[g];
}

// Gt[m][k] = G[k][m]  (so B[k][m]=G[k][m] is K-contiguous per output column m)
__global__ void k_transpose_G(const float* __restrict__ G, __bf16* __restrict__ Gt) {
  int g = blockIdx.x * 256 + threadIdx.x;       // 1M elems
  int m = g >> 10, k = g & 1023;
  Gt[g] = (__bf16)G[(size_t)k * NDIM + m];
}

// nvT[bt][d][n] = bf16(v[bt][n][d])   (transposed value matrix for attn@V)
__global__ void k_cvt_v(const float* __restrict__ vf, __bf16* __restrict__ nvT) {
  int g = blockIdx.x * 256 + threadIdx.x;       // 48*128*1024
  int bt  = g >> 17;
  int rem = g & ((DDIM * NDIM) - 1);
  int d = rem >> 10, n = rem & 1023;
  nvT[g] = (__bf16)vf[((size_t)bt * NDIM + n) * DDIM + d];
}

// ---------------- QKV projection ----------------
__global__ void k_qkv(const float* __restrict__ inp, const __bf16* __restrict__ Wb,
                      const float* __restrict__ bias, __bf16* __restrict__ qb,
                      __bf16* __restrict__ kb, float* __restrict__ vf) {
  const int lane = threadIdx.x & 31;
  const int wave = threadIdx.x >> 5;            // 4 waves, 6 col-tiles each
  const size_t row0 = (size_t)blockIdx.x * 16;

  v8f acc[6];
#pragma unroll
  for (int s = 0; s < 6; ++s) acc[s] = vzero8();

  for (int ks = 0; ks < DDIM; ks += 32) {
    v16bf a = fragA_f32(inp + row0 * DDIM + ks, DDIM, lane);
#pragma unroll
    for (int s = 0; s < 6; ++s) {
      const int jt = wave * 6 + s;
      v16bf b = fragB_b16(Wb + (size_t)(jt * 16) * DDIM + ks, DDIM, lane);
      acc[s] = wmma_bf16(a, b, acc[s]);
    }
  }
  const int n = lane & 15, mb = (lane >> 4) << 3;
  const float qscale = 0.08838834764831845f;    // 1/sqrt(128)
#pragma unroll
  for (int s = 0; s < 6; ++s) {
    const int col = (wave * 6 + s) * 16 + n;
    const float bc = bias[col];
#pragma unroll
    for (int j = 0; j < 8; ++j) {
      const size_t row = row0 + mb + j;
      const float v = acc[s][j] + bc;
      if (col < 128)      qb[row * DDIM + col]         = (__bf16)(v * qscale);
      else if (col < 256) kb[row * DDIM + (col - 128)] = (__bf16)v;
      else                vf[row * DDIM + (col - 256)] = v;
    }
  }
}

// ---------------- E = (q/sqrt(D)) @ k^T ----------------
__global__ void k_qk(const __bf16* __restrict__ qb, const __bf16* __restrict__ kb,
                     __bf16* __restrict__ Eb) {
  const int lane = threadIdx.x & 31, wave = threadIdx.x >> 5;
  const int gid = blockIdx.x;
  const int bt = gid >> 8;
  const int rb = (gid >> 2) & 63;
  const int cg = gid & 3;
  const int colbase = cg * 256 + wave * 64;
  const __bf16* qrow  = qb + ((size_t)bt * NDIM + rb * 16) * DDIM;
  const __bf16* kbase = kb + (size_t)bt * NDIM * DDIM;

  v8f acc[4];
#pragma unroll
  for (int s = 0; s < 4; ++s) acc[s] = vzero8();

  for (int ks = 0; ks < DDIM; ks += 32) {
    v16bf a = fragA_b16(qrow + ks, DDIM, lane);
#pragma unroll
    for (int s = 0; s < 4; ++s) {
      v16bf b = fragB_b16(kbase + (size_t)(colbase + s * 16) * DDIM + ks, DDIM, lane);
      acc[s] = wmma_bf16(a, b, acc[s]);
    }
  }
  const int n = lane & 15, mb = (lane >> 4) << 3;
  __bf16* erow = Eb + ((size_t)bt * NDIM + rb * 16) * NDIM;
#pragma unroll
  for (int s = 0; s < 4; ++s)
#pragma unroll
    for (int j = 0; j < 8; ++j)
      erow[(size_t)(mb + j) * NDIM + colbase + s * 16 + n] = (__bf16)acc[s][j];
}

// ---------------- attn = softmax(E @ G), in place over Eb ----------------
// 8 waves; E row-block (16x1024 bf16) staged in LDS (async DMA); 16x1024 f32
// accumulated in registers (8 tiles/wave); LDS reused for the f32 tile.
// Softmax fully parallel: 16 threads per row + LDS tree reductions.
__global__ void k_eg_softmax(__bf16* __restrict__ Eb, const __bf16* __restrict__ Gt) {
  __shared__ float Ss[16 * NDIM];               // 64KB; first half doubles as A-tile
  __shared__ float red[256];
  __shared__ float rstat[16];
  __bf16* As = (__bf16*)Ss;                     // 16x1024 bf16 (32KB)
  const int tid = threadIdx.x;
  const int bt = blockIdx.x >> 6, rb = blockIdx.x & 63;
  __bf16* Erows = Eb + ((size_t)bt * NDIM + rb * 16) * NDIM;

  stage16rows(Erows, As, tid);
  __syncthreads();

  const int lane = tid & 31, wave = tid >> 5;
  const int colbase = wave * 128;
  v8f acc[8];
#pragma unroll
  for (int s = 0; s < 8; ++s) acc[s] = vzero8();

  for (int kk = 0; kk < NDIM; kk += 32) {
    v16bf a = fragA_b16(As + kk, NDIM, lane);
#pragma unroll
    for (int s = 0; s < 8; ++s) {
      v16bf b = fragB_b16(Gt + (size_t)(colbase + s * 16) * NDIM + kk, NDIM, lane);
      acc[s] = wmma_bf16(a, b, acc[s]);
    }
  }
  __syncthreads();                              // A-tile dead, reuse LDS as f32
  const int n = lane & 15, mb = (lane >> 4) << 3;
#pragma unroll
  for (int s = 0; s < 8; ++s)
#pragma unroll
    for (int j = 0; j < 8; ++j)
      Ss[(mb + j) * NDIM + colbase + s * 16 + n] = acc[s][j];
  __syncthreads();

  // parallel row softmax: row r = tid>>4, strided segment seg = tid&15
  const int r = tid >> 4, seg = tid & 15;
  float mx = -3.4e38f;
#pragma unroll 4
  for (int i = 0; i < 64; ++i)
    mx = fmaxf(mx, Ss[r * NDIM + seg + 16 * i]);
  red[tid] = mx;
  __syncthreads();
  if (seg == 0) {
    float m = red[tid];
    for (int s = 1; s < 16; ++s) m = fmaxf(m, red[tid + s]);
    rstat[r] = m;
  }
  __syncthreads();
  const float rowmax = rstat[r];
  float psum = 0.f;
#pragma unroll 4
  for (int i = 0; i < 64; ++i) {
    const int m = seg + 16 * i;
    const float e = expf(Ss[r * NDIM + m] - rowmax);
    Ss[r * NDIM + m] = e;                       // cache exp value
    psum += e;
  }
  red[tid] = psum;
  __syncthreads();
  if (seg == 0) {
    float s0 = 0.f;
    for (int s = 0; s < 16; ++s) s0 += red[tid + s];
    rstat[r] = 1.f / s0;
  }
  __syncthreads();
  const float inv = rstat[r];
  __bf16* orow = Erows + (size_t)r * NDIM;
#pragma unroll 4
  for (int i = 0; i < 64; ++i) {
    const int m = seg + 16 * i;
    orow[m] = (__bf16)(Ss[r * NDIM + m] * inv);
  }
}

// ---------------- column sums of attn at t = XLAST ----------------
__global__ void k_colsum(const __bf16* __restrict__ attn, float* __restrict__ cs) {
  int g = blockIdx.x * 256 + threadIdx.x;       // 4*1024
  int b = g >> 10, m = g & 1023;
  const __bf16* p = attn + (size_t)(b * TDIM + XLAST) * NDIM * NDIM + m;
  float s = 0.f;
  for (int nq = 0; nq < NDIM; ++nq) s += (float)p[(size_t)nq * NDIM];
  cs[g] = s;
}

// y = index of 2nd-smallest column sum (top_k(-cs,2)[1], stable ties).
__global__ void k_select(const float* __restrict__ cs, int* __restrict__ yidx) {
  if (threadIdx.x != 0) return;
  const float* c = cs + blockIdx.x * NDIM;
  float v0 = 3.4e38f; int i0 = 0;
  for (int m = 0; m < NDIM; ++m) if (c[m] < v0) { v0 = c[m]; i0 = m; }
  float v1 = 3.4e38f; int i1 = 0;
  for (int m = 0; m < NDIM; ++m) if (m != i0 && c[m] < v1) { v1 = c[m]; i1 = m; }
  yidx[blockIdx.x] = i1;
}

// ---------------- mixup patch of new_value (front 78 channels) ----------------
__global__ void k_mixup(const float* __restrict__ vf, const int* __restrict__ yidx,
                        __bf16* __restrict__ nvT) {
  const int b = blockIdx.x, tid = threadIdx.x;
  __shared__ float mv[CFRONT];
  __shared__ int ysh;
  if (tid == 0) ysh = yidx[b];
  __syncthreads();
  const int y = ysh;

  if (tid < CFRONT) {
    float w[XLAST], ws = 0.f;
    for (int t = 0; t < XLAST; ++t) {
      const float d = (float)t - (float)(XLAST - 1);
      w[t] = expf(-(d * d) / (2.f * 10.f * 10.f));
      ws += w[t];
    }
    float acc = 0.f;
    for (int t = 0; t < XLAST; ++t)
      acc += w[t] * vf[((size_t)(b * TDIM + t) * NDIM + y) * DDIM + tid];
    mv[tid] = acc / ws;
  }
  __syncthreads();

  int y_start = y - BSST;        if (y_start < 0) y_start = 0;
  int y_end   = y + 2 * BSST;    if (y_end > NDIM) y_end = NDIM;

  if (y_end > NDIM - BSST) {     // tail branch: bs = 1, one column written
    const int count = y_end - y_start;
    for (int idx = tid; idx < TDIM * CFRONT; idx += 256) {
      const int t = idx / CFRONT, c = idx % CFRONT;
      float acc = 0.f;
      for (int col = y_start; col < y_end; ++col) {
        const float g = (t == XLAST && col == y)
            ? mv[c] : vf[((size_t)(b * TDIM + t) * NDIM + col) * DDIM + c];
        acc += g;
      }
      const float val = acc / (float)count;
      const unsigned char u8 = (unsigned char)(int)val;   // astype(uint8) trunc+wrap
      nvT[((size_t)(b * TDIM + t) * DDIM + c) * NDIM + y] = (__bf16)(float)u8;
    }
  } else {                       // normal branch: bs = 6
    const int j0 = y_start / BSST, j1 = y_end / BSST;
    const int count = j1 - j0;
    for (int idx = tid; idx < TDIM * BSST * CFRONT; idx += 256) {
      const int c = idx % CFRONT;
      const int i = (idx / CFRONT) % BSST;
      const int t = idx / (CFRONT * BSST);
      float acc = 0.f;
      for (int jj = 0; jj < count; ++jj) {
        int col = (j0 + jj) * BSST + i;
        if (col > NDIM - 1) col = NDIM - 1;
        const float g = (t == XLAST && col == y)
            ? mv[c] : vf[((size_t)(b * TDIM + t) * NDIM + col) * DDIM + c];
        acc += g;
      }
      const float val = acc / (float)count;
      const unsigned char u8 = (unsigned char)(int)val;
      nvT[((size_t)(b * TDIM + t) * DDIM + c) * NDIM + (y + i)] = (__bf16)(float)u8;
    }
  }
}

// ---------------- vi = attn @ new_value ----------------
__global__ void k_attnv(const __bf16* __restrict__ attn, const __bf16* __restrict__ nvT,
                        float* __restrict__ vif) {
  __shared__ __bf16 As[16 * NDIM];              // 32KB attn row-block
  const int tid = threadIdx.x;
  const int bt = blockIdx.x >> 6, rb = blockIdx.x & 63;
  const __bf16* Arows = attn + ((size_t)bt * NDIM + rb * 16) * NDIM;
  stage16rows(Arows, As, tid);
  __syncthreads();

  const int lane = tid & 31, wave = tid >> 5;
  const int d0 = wave * 16;                     // 8 waves x 16 cols = 128
  v8f acc = vzero8();
  for (int kk = 0; kk < NDIM; kk += 32) {
    v16bf a = fragA_b16(As + kk, NDIM, lane);
    v16bf b = fragB_b16(nvT + ((size_t)bt * DDIM + d0) * NDIM + kk, NDIM, lane);
    acc = wmma_bf16(a, b, acc);
  }
  const int n = lane & 15, mb = (lane >> 4) << 3;
  float* orow = vif + ((size_t)bt * NDIM + rb * 16) * DDIM;
#pragma unroll
  for (int j = 0; j < 8; ++j)
    orow[(size_t)(mb + j) * DDIM + d0 + n] = acc[j];
}

// ---------------- h = gelu(vi @ ff1^T + b1) ----------------
__global__ void k_ffn1(const float* __restrict__ vif, const __bf16* __restrict__ W,
                       const float* __restrict__ bias, __bf16* __restrict__ hb) {
  const int lane = threadIdx.x & 31, wave = threadIdx.x >> 5;
  const size_t row0 = (size_t)blockIdx.x * 16;
  v8f acc[2];
#pragma unroll
  for (int s = 0; s < 2; ++s) acc[s] = vzero8();

  for (int ks = 0; ks < DDIM; ks += 32) {
    v16bf a = fragA_f32(vif + row0 * DDIM + ks, DDIM, lane);
#pragma unroll
    for (int s = 0; s < 2; ++s) {
      const int jt = wave * 2 + s;
      v16bf b = fragB_b16(W + (size_t)(jt * 16) * DDIM + ks, DDIM, lane);
      acc[s] = wmma_bf16(a, b, acc[s]);
    }
  }
  const int n = lane & 15, mb = (lane >> 4) << 3;
#pragma unroll
  for (int s = 0; s < 2; ++s) {
    const int col = (wave * 2 + s) * 16 + n;
    const float bc = bias[col];
#pragma unroll
    for (int j = 0; j < 8; ++j) {
      const float v = acc[s][j] + bc;
      const float g = 0.5f * v * (1.f + erff(v * 0.70710678118654752f));
      hb[(row0 + mb + j) * DDIM + col] = (__bf16)g;
    }
  }
}

// ---------------- out = LN(h @ ff2^T + b2 + inp) * w + b ----------------
__global__ void k_ffn2_ln(const __bf16* __restrict__ hb, const __bf16* __restrict__ W,
                          const float* __restrict__ bias, const float* __restrict__ inp,
                          const float* __restrict__ lnw, const float* __restrict__ lnb,
                          float* __restrict__ out) {
  __shared__ float S[16 * DDIM];                // 8KB
  const int lane = threadIdx.x & 31, wave = threadIdx.x >> 5;
  const size_t row0 = (size_t)blockIdx.x * 16;
  v8f acc[2];
#pragma unroll
  for (int s = 0; s < 2; ++s) acc[s] = vzero8();

  for (int ks = 0; ks < DDIM; ks += 32) {
    v16bf a = fragA_b16(hb + row0 * DDIM + ks, DDIM, lane);
#pragma unroll
    for (int s = 0; s < 2; ++s) {
      const int jt = wave * 2 + s;
      v16bf b = fragB_b16(W + (size_t)(jt * 16) * DDIM + ks, DDIM, lane);
      acc[s] = wmma_bf16(a, b, acc[s]);
    }
  }
  const int n = lane & 15, mb = (lane >> 4) << 3;
#pragma unroll
  for (int s = 0; s < 2; ++s) {
    const int col = (wave * 2 + s) * 16 + n;
    const float bc = bias[col];
#pragma unroll
    for (int j = 0; j < 8; ++j) {
      const int m = mb + j;
      S[m * DDIM + col] = acc[s][j] + bc + inp[(row0 + m) * DDIM + col];
    }
  }
  __syncthreads();
  if (threadIdx.x < 16) {
    const int r = threadIdx.x;
    float mu = 0.f;
    for (int c = 0; c < DDIM; ++c) mu += S[r * DDIM + c];
    mu *= (1.f / DDIM);
    float var = 0.f;
    for (int c = 0; c < DDIM; ++c) { const float d = S[r * DDIM + c] - mu; var += d * d; }
    var *= (1.f / DDIM);
    const float rs = rsqrtf(var + 1e-5f);
    float* orow = out + (row0 + r) * DDIM;
    for (int c = 0; c < DDIM; ++c)
      orow[c] = (S[r * DDIM + c] - mu) * rs * lnw[c] + lnb[c];
  }
}

// ---------------- host launcher ----------------

extern "C" void kernel_launch(void* const* d_in, const int* in_sizes, int n_in,
                              void* d_out, int out_size, void* d_ws, size_t ws_size,
                              hipStream_t stream) {
  (void)in_sizes; (void)n_in; (void)out_size; (void)ws_size;
  const float* inp   = (const float*)d_in[0];
  const float* G     = (const float*)d_in[1];
  const float* xff_w = (const float*)d_in[2];
  const float* xff_b = (const float*)d_in[3];
  const float* ff1_w = (const float*)d_in[4];
  const float* ff1_b = (const float*)d_in[5];
  const float* ff2_w = (const float*)d_in[6];
  const float* ff2_b = (const float*)d_in[7];
  const float* ln_w  = (const float*)d_in[8];
  const float* ln_b  = (const float*)d_in[9];
  float* out = (float*)d_out;
  char* ws = (char*)d_ws;

  __bf16* qb  = (__bf16*)(ws + 0);           // 48*1024*128 bf16   (12.6 MB)
  __bf16* kb  = (__bf16*)(ws + 12582912);    // 48*1024*128 bf16
  float*  vf  = (float*) (ws + 25165824);    // 48*1024*128 f32    (25.2 MB)
  __bf16* Eb  = (__bf16*)(ws + 50331648);    // 48*1024*1024 bf16  (100.7 MB, reused as attn)
  __bf16* nvT = (__bf16*)(ws + 150994944);   // 48*128*1024 bf16   (transposed new_value)
  float*  vif = (float*) (ws + 163577856);   // 48*1024*128 f32
  __bf16* hb  = (__bf16*)(ws + 188743680);   // 48*1024*128 bf16
  __bf16* Gt  = (__bf16*)(ws + 201326592);   // 1024*1024 bf16 (transposed G)
  __bf16* xwb = (__bf16*)(ws + 203423744);   // 384*128 bf16
  __bf16* f1b = (__bf16*)(ws + 203522048);   // 128*128 bf16
  __bf16* f2b = (__bf16*)(ws + 203554816);   // 128*128 bf16
  float*  cs  = (float*) (ws + 203587584);   // 4*1024 f32
  int*    yix = (int*)   (ws + 203603968);   // 4 ints

  k_cvt_bf16   <<<192,   256, 0, stream>>>(xff_w, xwb, 49152);
  k_cvt_bf16   <<<64,    256, 0, stream>>>(ff1_w, f1b, 16384);
  k_cvt_bf16   <<<64,    256, 0, stream>>>(ff2_w, f2b, 16384);
  k_transpose_G<<<4096,  256, 0, stream>>>(G, Gt);
  k_qkv        <<<3072,  128, 0, stream>>>(inp, xwb, xff_b, qb, kb, vf);
  k_qk         <<<12288, 128, 0, stream>>>(qb, kb, Eb);
  k_eg_softmax <<<3072,  256, 0, stream>>>(Eb, Gt);
  k_colsum     <<<16,    256, 0, stream>>>(Eb, cs);
  k_select     <<<4,     32,  0, stream>>>(cs, yix);
  k_cvt_v      <<<24576, 256, 0, stream>>>(vf, nvT);
  k_mixup      <<<4,     256, 0, stream>>>(vf, yix, nvT);
  k_attnv      <<<3072,  256, 0, stream>>>(Eb, nvT, vif);
  k_ffn1       <<<3072,  128, 0, stream>>>(vif, f1b, ff1_b, hb);
  k_ffn2_ln    <<<3072,  128, 0, stream>>>(hb, f2b, ff2_b, inp, ln_w, ln_b, out);
}
